// GPT2Attention_62947040690425
// MI455X (gfx1250) — matrix operand order repaired and checked by reading, not compiled, not run on hardware
//
#include <hip/hip_runtime.h>

// GPT-2 attention block for MI455X (gfx1250, wave32, WMMA 16x16x32 bf16).
// Pipeline: f32->bf16 convert, QKV GEMM (64x64 wave tiles, wmma), flash
// attention (async double-buffered K/V staging to LDS, wmma), output-proj
// GEMM (wmma, fp32 out). Intermediates live in d_ws (~48 MB, L2-resident).

#define EMBD  1024
#define NHEAD 16
#define HDIM  64
#define BATCH 2
#define SEQ   2048
#define QKV3  3072
#define NEGV  (-10000.0f)

typedef __attribute__((ext_vector_type(16))) __bf16 v16bf;
typedef __attribute__((ext_vector_type(8)))  __bf16 v8bf;
typedef __attribute__((ext_vector_type(8)))  float  v8f;
typedef __attribute__((ext_vector_type(4)))  int    v4i;

typedef __attribute__((address_space(1))) v4i* gv4i_ptr;   // global int4*
typedef __attribute__((address_space(3))) v4i* lv4i_ptr;   // LDS int4*

static __device__ __forceinline__ v16bf mk16(v8bf lo, v8bf hi) {
  v16bf r;
#pragma unroll
  for (int i = 0; i < 8; ++i) { r[i] = lo[i]; r[i + 8] = hi[i]; }
  return r;
}

static __device__ __forceinline__ v8f wmma_bf16(v16bf a, v16bf b, v8f c) {
  // 8 args: (neg_a, A, neg_b, B, c_mod, C, reuse_a, reuse_b)
  return __builtin_amdgcn_wmma_f32_16x16x32_bf16(false, a, false, b,
                                                 (short)0, c, false, false);
}

// 8 elements per thread, 16B bf16 stores.
__global__ void cvt_f32_bf16(const float* __restrict__ x, __bf16* __restrict__ y, int n8) {
  int i = blockIdx.x * blockDim.x + threadIdx.x;
  if (i < n8) {
    v8bf r;
#pragma unroll
    for (int k = 0; k < 8; ++k) r[k] = (__bf16)x[i * 8 + k];
    *(v8bf*)(y + (size_t)i * 8) = r;
  }
}

// C[M,N] = A[M,K] @ Bw[N,K]^T + bias ; per-wave 64(M) x 64(N) tile.
// 4 A-frags + 4 B-frags -> 16 wmma per 32-wide k-step.
template <bool OUT_BF16>
__global__ __launch_bounds__(256) void gemm_bf16(const __bf16* __restrict__ A,
                                                 const __bf16* __restrict__ Bw,
                                                 const float* __restrict__ bias,
                                                 void* __restrict__ Cout,
                                                 int M, int N, int K) {
  const int w = threadIdx.x >> 5;
  const int lane = threadIdx.x & 31;
  const int hs = lane >> 4, ln = lane & 15;
  const int Nt4 = N >> 6;
  const int total = (M >> 6) * Nt4;
  const int waveId = blockIdx.x * 8 + w;
  if (waveId >= total) return;  // wave-uniform exit
  const int mt = waveId / Nt4;
  const int n4 = waveId % Nt4;

  const __bf16* abase = A + (size_t)((mt << 6) + ln) * K;
  v8f c[4][4] = {};  // [m-subtile][n-subtile]

  for (int k0 = 0; k0 < K; k0 += 32) {
    v16bf af[4];
#pragma unroll
    for (int t = 0; t < 4; ++t) {
      // A fragment: lane row = 16t + ln, 16 contiguous K at k0 + 16*hs
      const __bf16* ar = abase + (size_t)(t << 4) * K + k0;
      af[t] = *(const v16bf*)(ar + 16 * hs);
    }
#pragma unroll
    for (int f = 0; f < 4; ++f) {
      const int col = (n4 << 6) + (f << 4) + ln;
      // B fragment: lane col = ln, 16 contiguous K at k0 + 16*hs
      const v16bf bfr = *(const v16bf*)(Bw + (size_t)col * K + k0 + 16 * hs);
#pragma unroll
      for (int t = 0; t < 4; ++t) c[t][f] = wmma_bf16(af[t], bfr, c[t][f]);
    }
  }

#pragma unroll
  for (int f = 0; f < 4; ++f) {
    const int col = (n4 << 6) + (f << 4) + ln;
    const float bb = bias[col];
#pragma unroll
    for (int t = 0; t < 4; ++t)
#pragma unroll
      for (int r = 0; r < 8; ++r) {
        const int orow = (mt << 6) + (t << 4) + r + (hs << 3);
        const float v = c[t][f][r] + bb;
        if (OUT_BF16)
          ((__bf16*)Cout)[(size_t)orow * N + col] = (__bf16)v;
        else
          ((float*)Cout)[(size_t)orow * N + col] = v;
      }
  }
}

// Flash attention: block = 8 waves = 128 q rows for one (b,h). 32-key steps,
// double-buffered async global->LDS staging of K/V tiles.
__global__ __launch_bounds__(256) void attn_flash(const __bf16* __restrict__ qkv,
                                                  const int* __restrict__ amask,
                                                  __bf16* __restrict__ ctx) {
  __shared__ __bf16 Kt[2][32 * 64];
  __shared__ __bf16 Vt[2][32 * 64];
  __shared__ int    mkv[2][32];
  __shared__ __bf16 Pt[8][16 * 32];

  const int tid = threadIdx.x;
  const int w = tid >> 5, lane = tid & 31;
  const int hs = lane >> 4, ln = lane & 15;
  const int qt = blockIdx.x, h = blockIdx.y, b = blockIdx.z;
  const int qbase = qt * 128;
  const int qr0 = qbase + w * 16;

  // Q A-fragments (d = 0..31 and 32..63)
  const __bf16* qp = qkv + (size_t)(b * SEQ + qr0 + ln) * QKV3 + h * HDIM;
  const v16bf qa0 = *(const v16bf*)(qp + 16 * hs);
  const v16bf qa1 = *(const v16bf*)(qp + 32 + 16 * hs);

  float mrow[8], lrow[8];
  v8f o[4] = {};
#pragma unroll
  for (int r = 0; r < 8; ++r) { mrow[r] = -1e30f; lrow[r] = 0.0f; }

  // per-thread staging slice: 8 bf16 (16B) of K and of V per tile
  const int sj = tid >> 3;            // key row 0..31
  const int sd = (tid & 7) << 3;      // d col {0,8,...,56}
  const int ntiles = (qbase + 128) >> 5;

  // issue async fill of tile jt into buffer buf
  auto issue_tile = [&](int jt, int buf) {
    const size_t ro = (size_t)(b * SEQ + (jt << 5) + sj) * QKV3 + h * HDIM + sd;
    __builtin_amdgcn_global_load_async_to_lds_b128(
        (gv4i_ptr)(qkv + ro + EMBD),
        (lv4i_ptr)&Kt[buf][sj * 64 + sd], 0, 0);
    __builtin_amdgcn_global_load_async_to_lds_b128(
        (gv4i_ptr)(qkv + ro + 2 * EMBD),
        (lv4i_ptr)&Vt[buf][sj * 64 + sd], 0, 0);
    if (tid < 32) mkv[buf][tid] = (amask[b * SEQ + (jt << 5) + tid] == 0);
  };

  issue_tile(0, 0);

  for (int jt = 0; jt < ntiles; ++jt) {
    const int jbase = jt << 5;
    const int cur = jt & 1;
    asm volatile("s_wait_asynccnt 0x0" ::: "memory");
    __syncthreads();  // buffer `cur` ready for all waves
    if (jt + 1 < ntiles) issue_tile(jt + 1, cur ^ 1);  // overlap next fill

    if (jbase <= qr0 + 15) {  // wave-uniform causal skip
      v8f s0 = {}, s1 = {};
      {
        const __bf16* kr0 = &Kt[cur][ln * 64 + 16 * hs];         // cols 0..15
        s0 = wmma_bf16(qa0, *(const v16bf*)(kr0), s0);
        s0 = wmma_bf16(qa1, *(const v16bf*)(kr0 + 32), s0);
        const __bf16* kr1 = &Kt[cur][(16 + ln) * 64 + 16 * hs];  // cols 16..31
        s1 = wmma_bf16(qa0, *(const v16bf*)(kr1), s1);
        s1 = wmma_bf16(qa1, *(const v16bf*)(kr1 + 32), s1);
      }

      float alpha[8];
      const int kg0 = jbase + ln, kg1 = jbase + 16 + ln;
      const int m0 = mkv[cur][ln], m1 = mkv[cur][16 + ln];
#pragma unroll
      for (int r = 0; r < 8; ++r) {
        const int q = qr0 + r + (hs << 3);
        float a0 = s0[r] * 0.125f;  // HDIM^-0.5
        float a1 = s1[r] * 0.125f;
        if (kg0 > q || m0) a0 = NEGV;
        if (kg1 > q || m1) a1 = NEGV;
        float v = fmaxf(a0, a1);
        v = fmaxf(v, __shfl_xor(v, 1, 16));
        v = fmaxf(v, __shfl_xor(v, 2, 16));
        v = fmaxf(v, __shfl_xor(v, 4, 16));
        v = fmaxf(v, __shfl_xor(v, 8, 16));
        const float mnew = fmaxf(mrow[r], v);
        const float al = __expf(mrow[r] - mnew);
        mrow[r] = mnew;
        alpha[r] = al;
        const float p0 = __expf(a0 - mnew);
        const float p1 = __expf(a1 - mnew);
        float ps = p0 + p1;
        ps += __shfl_xor(ps, 1, 16);
        ps += __shfl_xor(ps, 2, 16);
        ps += __shfl_xor(ps, 4, 16);
        ps += __shfl_xor(ps, 8, 16);
        lrow[r] = lrow[r] * al + ps;
        // stash P tile (C-layout -> row-major bf16) in per-wave LDS
        Pt[w][(r + (hs << 3)) * 32 + ln] = (__bf16)p0;
        Pt[w][(r + (hs << 3)) * 32 + 16 + ln] = (__bf16)p1;
      }
#pragma unroll
      for (int c = 0; c < 4; ++c)
#pragma unroll
        for (int r = 0; r < 8; ++r) o[c][r] *= alpha[r];

      // P as A-fragment (row = ln, 16 contiguous K at 16*hs)
      const v16bf pa = *(const v16bf*)&Pt[w][ln * 32 + 16 * hs];
      // V as B-fragments: lane col d = 16c + ln, K = 16*hs + i (column gather)
#pragma unroll
      for (int c = 0; c < 4; ++c) {
        const int dcol = (c << 4) + ln;
        v16bf vb;
#pragma unroll
        for (int i = 0; i < 16; ++i) vb[i] = Vt[cur][(16 * hs + i) * 64 + dcol];
        o[c] = wmma_bf16(pa, vb, o[c]);
      }
    }
  }

  float inv[8];
#pragma unroll
  for (int r = 0; r < 8; ++r) inv[r] = 1.0f / lrow[r];
#pragma unroll
  for (int c = 0; c < 4; ++c)
#pragma unroll
    for (int r = 0; r < 8; ++r) {
      const int q = qr0 + r + (hs << 3);
      const int d = (c << 4) + ln;
      ctx[(size_t)(b * SEQ + q) * EMBD + h * HDIM + d] = (__bf16)(o[c][r] * inv[r]);
    }
}

extern "C" void kernel_launch(void* const* d_in, const int* in_sizes, int n_in,
                              void* d_out, int out_size, void* d_ws, size_t ws_size,
                              hipStream_t stream) {
  const float* hs = (const float*)d_in[0];
  const int*   am = (const int*)d_in[1];
  const float* Wa = (const float*)d_in[2];
  const float* ba = (const float*)d_in[3];
  const float* Wp = (const float*)d_in[4];
  const float* bp = (const float*)d_in[5];
  (void)in_sizes; (void)n_in; (void)out_size; (void)ws_size;

  // Workspace layout (bf16 elements), total ~48 MB:
  __bf16* Xh   = (__bf16*)d_ws;                 // 4096*1024
  __bf16* Wah  = Xh   + (size_t)4096 * 1024;    // 3072*1024
  __bf16* Wph  = Wah  + (size_t)3072 * 1024;    // 1024*1024
  __bf16* qkvh = Wph  + (size_t)1024 * 1024;    // 4096*3072
  __bf16* ctxh = qkvh + (size_t)4096 * 3072;    // 4096*1024

  int n8 = (4096 * 1024) / 8;
  cvt_f32_bf16<<<(n8 + 255) / 256, 256, 0, stream>>>(hs, Xh, n8);
  n8 = (3072 * 1024) / 8;
  cvt_f32_bf16<<<(n8 + 255) / 256, 256, 0, stream>>>(Wa, Wah, n8);
  n8 = (1024 * 1024) / 8;
  cvt_f32_bf16<<<(n8 + 255) / 256, 256, 0, stream>>>(Wp, Wph, n8);

  {  // qkv = X @ W_attn^T + b_attn  (bf16 out)
    const int M = 4096, N = 3072, K = 1024;
    const int waves = (M / 64) * (N / 64);
    gemm_bf16<true><<<(waves + 7) / 8, 256, 0, stream>>>(Xh, Wah, ba, (void*)qkvh, M, N, K);
  }

  attn_flash<<<dim3(SEQ / 128, NHEAD, BATCH), 256, 0, stream>>>(qkvh, am, ctxh);

  {  // out = ctx @ W_proj^T + b_proj  (fp32 out)
    const int M = 4096, N = 1024, K = 1024;
    const int waves = (M / 64) * (N / 64);
    gemm_bf16<false><<<(waves + 7) / 8, 256, 0, stream>>>(ctxh, Wph, bp, d_out, M, N, K);
  }
}
